// EwaldPotential_79422535238417
// MI455X (gfx1250) — matrix operand order repaired and checked by reading, not compiled, not run on hardware
//
#include <hip/hip_runtime.h>
#include <math.h>

typedef __attribute__((ext_vector_type(16))) _Float16 v16h;
typedef __attribute__((ext_vector_type(8)))  _Float16 v8h;
typedef __attribute__((ext_vector_type(8)))  float    v8f;

#define D_CH 128
#define N_AT 8192

__device__ __forceinline__ v8f wmma32(v16h a, v16h b, v8f c) {
  // (neg_a, A, neg_b, B, c_mod, C, reuse_a, reuse_b)
  return __builtin_amdgcn_wmma_f32_16x16x32_f16(false, a, false, b, (short)0, c,
                                                false, false);
}

__device__ __forceinline__ v16h cat16(v8h lo, v8h hi) {
  return __builtin_shufflevector(lo, hi, 0, 1, 2, 3, 4, 5, 6, 7, 8, 9, 10, 11,
                                 12, 13, 14, 15);
}

// ---------------------------------------------------------------------------
// Stage A: structure-factor potentials.
//   k_pot/v_pot[k,d] = sum_n exp(i*2pi r_n . k_fwd_k) * {k,v}[n,d]
// Block = 16 k-rows x full D (8 waves, one 16-wide d-tile each).
// Superstep = 64 atoms; the cos/sin A-tiles are computed cooperatively by all
// 8 waves into LDS (8 transcendentals per lane per superstep instead of 64),
// then every wave consumes them for its own d-tile WMMAs.
// Outputs (f16): kabs[K,D] = |k_pot|, vreT/vimT[D,Kpad] = v_pot transposed.
// ---------------------------------------------------------------------------
__global__ __launch_bounds__(256) void ewald_pot_kernel(
    const float* __restrict__ kvec, const float* __restrict__ vvec,
    const float* __restrict__ pos, const float* __restrict__ cell,
    const int* __restrict__ kfwd, int K, int Kpad,
    _Float16* __restrict__ kabs, _Float16* __restrict__ vreT,
    _Float16* __restrict__ vimT) {
  __shared__ __align__(32) _Float16 skv[D_CH * 72];   // [d][n], stride 72
  __shared__ __align__(32) _Float16 svv[D_CH * 72];
  __shared__ __align__(32) _Float16 sAc[2][16 * 32];  // cos tiles, [row][n]
  __shared__ __align__(32) _Float16 sAs[2][16 * 32];  // sin tiles
  __shared__ __align__(16) float srf[64 * 3];

  const int tid = threadIdx.x;
  const int lane = tid & 31;
  const int wave = tid >> 5;
  const int hi = lane >> 4;  // 0 / 1 lane half
  const int l16 = lane & 15;

  const int kt = blockIdx.x * 16;  // base k-row of this block
  const int dt = wave * 16;        // base d-col of this wave

  const float bx = cell[0], by = cell[4], bz = cell[8];

  // per-lane k-vector (A-matrix row = l16); padded rows masked in epilogue
  const int krow = kt + l16;
  const int kc = (krow < K) ? krow : 0;
  const float gx = (float)kfwd[kc * 3 + 0];
  const float gy = (float)kfwd[kc * 3 + 1];
  const float gz = (float)kfwd[kc * 3 + 2];

  // cooperative-trig work split: chunk (0/1), cos vs sin, n-half (0/1)
  const int tg_c = wave & 1;
  const int tg_sin = (wave >> 1) & 1;
  const int tg_nb = (wave >> 2) * 16 + hi * 8;  // base n within chunk

  const int dcol = dt + l16;
  const int roff = hi * 16;

  v8f akre = {}, akim = {}, avre = {}, avim = {};

  for (int n0 = 0; n0 < N_AT; n0 += 64) {
    // ---- stage fractional coordinates of 64 atoms ----
    if (tid < 192) {
      int n = tid / 3, c = tid - 3 * n;
      float b = (c == 0) ? bx : ((c == 1) ? by : bz);
      srf[tid] = pos[(size_t)(n0 + n) * 3 + c] / b;
    }
    __syncthreads();

    // ---- stage k_vector / v_vector chunk transposed to [d][n] as f16 ----
    for (int i = tid; i < 64 * D_CH; i += 256) {
      int n = i >> 7, d = i & 127;
      skv[d * 72 + n] = (_Float16)kvec[(size_t)(n0 + n) * D_CH + d];
      svv[d * 72 + n] = (_Float16)vvec[(size_t)(n0 + n) * D_CH + d];
    }

    // ---- cooperative trig: 8 values per lane, whole block covers both
    //      16x32 cos and sin tiles for the two 32-atom chunks ----
    {
      v8h tv;
#pragma unroll
      for (int j = 0; j < 8; ++j) {
        int n = tg_c * 32 + tg_nb + j;  // atom within superstep
        float t = gx * srf[n * 3 + 0] + gy * srf[n * 3 + 1] + gz * srf[n * 3 + 2];
        float r = tg_sin ? __builtin_amdgcn_sinf(t) : __builtin_amdgcn_cosf(t);
        tv[j] = (_Float16)r;
      }
      _Float16* dst = (tg_sin ? sAs[tg_c] : sAc[tg_c]) + l16 * 32 + tg_nb;
      *(v8h*)dst = tv;
    }
    __syncthreads();

    // ---- 8 WMMAs per superstep from shared A tiles + this wave's B tiles ----
#pragma unroll
    for (int c = 0; c < 2; ++c) {
      v16h Ac = cat16(*(const v8h*)&sAc[c][l16 * 32 + hi * 16],
                      *(const v8h*)&sAc[c][l16 * 32 + hi * 16 + 8]);
      v16h As = cat16(*(const v8h*)&sAs[c][l16 * 32 + hi * 16],
                      *(const v8h*)&sAs[c][l16 * 32 + hi * 16 + 8]);
      const int off = dcol * 72 + c * 32 + roff;
      v16h Bk = cat16(*(const v8h*)&skv[off], *(const v8h*)&skv[off + 8]);
      v16h Bv = cat16(*(const v8h*)&svv[off], *(const v8h*)&svv[off + 8]);
      akre = wmma32(Ac, Bk, akre);
      akim = wmma32(As, Bk, akim);
      avre = wmma32(Ac, Bv, avre);
      avim = wmma32(As, Bv, avim);
    }
    __syncthreads();
  }

#pragma unroll
  for (int j = 0; j < 8; ++j) {
    int kr = kt + hi * 8 + j;  // C row
    int d = dt + l16;          // C col
    bool valid = kr < K;
    float re = akre[j], im = akim[j];
    float mag = valid ? sqrtf(re * re + im * im) : 0.f;
    kabs[(size_t)kr * D_CH + d] = (_Float16)mag;
    vreT[(size_t)d * Kpad + kr] = (_Float16)(valid ? avre[j] : 0.f);
    vimT[(size_t)d * Kpad + kr] = (_Float16)(valid ? avim[j] : 0.f);
  }
}

// ---------------------------------------------------------------------------
// Stage B: fused |q|@|k_pot|^T -> online softmax -> Re(inverse transform).
// One 16-atom tile per wave, streaming over k in 32-wide steps.
// ---------------------------------------------------------------------------
__global__ __launch_bounds__(256) void ewald_attn_kernel(
    const float* __restrict__ q, const float* __restrict__ pos,
    const float* __restrict__ cell, const int* __restrict__ kinv, int K,
    int Kpad, const _Float16* __restrict__ kabs,
    const _Float16* __restrict__ vreT, const _Float16* __restrict__ vimT,
    float* __restrict__ out) {
  __shared__ __align__(32) _Float16 sP[8][2][16 * 32];  // per-wave P.cos/P.sin

  const int tid = threadIdx.x, lane = tid & 31, wave = tid >> 5;
  const int hi = lane >> 4, l16 = lane & 15;
  const int n0 = (blockIdx.x * 8 + wave) * 16;

  const float bx = cell[0], by = cell[4], bz = cell[8];

  // |q| A-fragments for the aw GEMM (4 d-chunks of 32)
  v16h Aq[4];
  {
    const float* qr = q + (size_t)(n0 + l16) * D_CH;
#pragma unroll
    for (int dc = 0; dc < 4; ++dc)
#pragma unroll
      for (int h = 0; h < 16; ++h) {
        int d = dc * 32 + hi * 16 + h;
        Aq[dc][h] = (_Float16)fabsf(qr[d]);
      }
  }

  // fractional coordinates of this lane's 8 output rows
  float rx[8], ry[8], rz[8];
#pragma unroll
  for (int j = 0; j < 8; ++j) {
    int n = n0 + hi * 8 + j;
    rx[j] = pos[(size_t)n * 3 + 0] / bx;
    ry[j] = pos[(size_t)n * 3 + 1] / by;
    rz[j] = pos[(size_t)n * 3 + 2] / bz;
  }

  v8f acc[8];
  v8f vzero = {};
#pragma unroll
  for (int t = 0; t < 8; ++t) acc[t] = vzero;
  float m[8], lsum[8];
#pragma unroll
  for (int j = 0; j < 8; ++j) {
    m[j] = -1e30f;
    lsum[j] = 0.f;
  }

  const int roff = hi * 16;
  _Float16* pc = &sP[wave][0][0];
  _Float16* ps = &sP[wave][1][0];

  for (int kt0 = 0; kt0 < Kpad; kt0 += 32) {
    const int kcol0 = kt0 + l16;
    const int kcol1 = kt0 + 16 + l16;

    // aw tiles: two 16-wide column groups
    v8f aw0 = vzero, aw1 = vzero;
#pragma unroll
    for (int dc = 0; dc < 4; ++dc) {
      const _Float16* p0 = kabs + (size_t)kcol0 * D_CH + dc * 32 + roff;
      const _Float16* p1 = kabs + (size_t)kcol1 * D_CH + dc * 32 + roff;
      v16h B0 = cat16(*(const v8h*)p0, *(const v8h*)(p0 + 8));
      v16h B1 = cat16(*(const v8h*)p1, *(const v8h*)(p1 + 8));
      aw0 = wmma32(Aq[dc], B0, aw0);
      aw1 = wmma32(Aq[dc], B1, aw1);
    }

    const bool ok0 = kcol0 < K, ok1 = kcol1 < K;
    const int kc0 = ok0 ? kcol0 : 0, kc1 = ok1 ? kcol1 : 0;
    const float g0x = (float)kinv[kc0 * 3 + 0], g0y = (float)kinv[kc0 * 3 + 1],
                g0z = (float)kinv[kc0 * 3 + 2];
    const float g1x = (float)kinv[kc1 * 3 + 0], g1y = (float)kinv[kc1 * 3 + 1],
                g1z = (float)kinv[kc1 * 3 + 2];

    // online softmax per row + P*cos_i / P*sin_i into LDS (C-layout -> tile)
#pragma unroll
    for (int j = 0; j < 8; ++j) {
      float a0 = ok0 ? aw0[j] : -1e30f;
      float a1 = ok1 ? aw1[j] : -1e30f;
      float tmax = fmaxf(a0, a1);
#pragma unroll
      for (int msk = 8; msk >= 1; msk >>= 1)
        tmax = fmaxf(tmax, __shfl_xor(tmax, msk, 32));
      float mnew = fmaxf(m[j], tmax);
      float scale = __expf(m[j] - mnew);
      float p0 = __expf(a0 - mnew);
      float p1 = __expf(a1 - mnew);
      float rsum = p0 + p1;
#pragma unroll
      for (int msk = 8; msk >= 1; msk >>= 1) rsum += __shfl_xor(rsum, msk, 32);
      lsum[j] = lsum[j] * scale + rsum;
      m[j] = mnew;
#pragma unroll
      for (int t = 0; t < 8; ++t) acc[t][j] *= scale;

      float t0 = g0x * rx[j] + g0y * ry[j] + g0z * rz[j];  // phase / 2pi
      float t1 = g1x * rx[j] + g1y * ry[j] + g1z * rz[j];
      int row = hi * 8 + j;
      pc[row * 32 + l16] = (_Float16)(p0 * __builtin_amdgcn_cosf(t0));
      ps[row * 32 + l16] = (_Float16)(p0 * __builtin_amdgcn_sinf(t0));
      pc[row * 32 + 16 + l16] = (_Float16)(p1 * __builtin_amdgcn_cosf(t1));
      ps[row * 32 + 16 + l16] = (_Float16)(p1 * __builtin_amdgcn_sinf(t1));
    }

    asm volatile("s_wait_dscnt 0" ::: "memory");  // intra-wave LDS RAW

    // reload P tiles in A-fragment layout
    v16h Pc, Ps;
    {
      int r = l16, coff = hi * 16;
      Pc = cat16(*(const v8h*)&pc[r * 32 + coff],
                 *(const v8h*)&pc[r * 32 + coff + 8]);
      Ps = cat16(*(const v8h*)&ps[r * 32 + coff],
                 *(const v8h*)&ps[r * 32 + coff + 8]);
    }

    // out += Pc @ v_re + Ps @ v_im  (8 d-tiles)
#pragma unroll
    for (int t = 0; t < 8; ++t) {
      const _Float16* pr = vreT + (size_t)(t * 16 + l16) * Kpad + kt0 + roff;
      const _Float16* pi = vimT + (size_t)(t * 16 + l16) * Kpad + kt0 + roff;
      v16h Br = cat16(*(const v8h*)pr, *(const v8h*)(pr + 8));
      v16h Bi = cat16(*(const v8h*)pi, *(const v8h*)(pi + 8));
      acc[t] = wmma32(Pc, Br, acc[t]);
      acc[t] = wmma32(Ps, Bi, acc[t]);
    }
    asm volatile("s_wait_dscnt 0" ::: "memory");  // LDS WAR before next step
  }

  // normalize by softmax denominator and store
#pragma unroll
  for (int j = 0; j < 8; ++j) {
    float inv = 1.f / lsum[j];
    int n = n0 + hi * 8 + j;
#pragma unroll
    for (int t = 0; t < 8; ++t)
      out[(size_t)n * D_CH + t * 16 + l16] = acc[t][j] * inv;
  }
}

extern "C" void kernel_launch(void* const* d_in, const int* in_sizes, int n_in,
                              void* d_out, int out_size, void* d_ws,
                              size_t ws_size, hipStream_t stream) {
  (void)n_in;
  (void)out_size;
  (void)ws_size;
  const float* qv = (const float*)d_in[0];
  const float* kv = (const float*)d_in[1];
  const float* vv = (const float*)d_in[2];
  const float* pos = (const float*)d_in[3];
  const float* cell = (const float*)d_in[4];
  const int* kfwd = (const int*)d_in[6];
  const int* kinv = (const int*)d_in[7];

  const int K = in_sizes[6] / 3;
  const int Kpad = (K + 31) & ~31;

  char* ws = (char*)d_ws;
  _Float16* kabs = (_Float16*)ws;                              // [K,D]
  _Float16* vreT = (_Float16*)(ws + (size_t)Kpad * D_CH * 2);  // [D,Kpad]
  _Float16* vimT = (_Float16*)(ws + (size_t)Kpad * D_CH * 4);  // [D,Kpad]

  ewald_pot_kernel<<<Kpad / 16, 256, 0, stream>>>(kv, vv, pos, cell, kfwd, K,
                                                  Kpad, kabs, vreT, vimT);
  ewald_attn_kernel<<<N_AT / 128, 256, 0, stream>>>(
      qv, pos, cell, kinv, K, Kpad, kabs, vreT, vimT, (float*)d_out);
}
